// Attention_83090437308676
// MI455X (gfx1250) — matrix operand order repaired and checked by reading, not compiled, not run on hardware
//
#include <hip/hip_runtime.h>
#include <hip/hip_bf16.h>

#define SEQ   4096
#define DM    2048
#define DH    128
#define DKV   512
#define NCOLS 3072

typedef __attribute__((ext_vector_type(16))) __bf16 v16bf;
typedef __attribute__((ext_vector_type(8)))  float  v8f;
typedef __attribute__((ext_vector_type(4)))  int    v4i;

union BF16x16 { unsigned short u[16]; v16bf v; };

#if defined(__HIP_DEVICE_COMPILE__) && __has_builtin(__builtin_amdgcn_global_load_async_to_lds_b128)
#define USE_ASYNC 1
#define AS1 __attribute__((address_space(1)))
#define AS3 __attribute__((address_space(3)))
#else
#define USE_ASYNC 0
#endif

__device__ __forceinline__ unsigned short f2bf(float f) {
  unsigned int x = __builtin_bit_cast(unsigned int, f);
  unsigned int r = x + 0x7FFFu + ((x >> 16) & 1u);   // round-to-nearest-even
  return (unsigned short)(r >> 16);
}
__device__ __forceinline__ float bf2f(unsigned short u) {
  return __builtin_bit_cast(float, (unsigned int)u << 16);
}

// 16-byte global -> LDS copy; async (ASYNCcnt-tracked, no VGPR round trip) when
// the CDNA5 builtin is available, plain vector copy otherwise.
__device__ __forceinline__ void cp128(unsigned short* l, const unsigned short* g) {
#if USE_ASYNC
  __builtin_amdgcn_global_load_async_to_lds_b128(
      (AS1 v4i*)g, (AS3 v4i*)l, 0, 0);
#else
  *reinterpret_cast<uint4*>(l) = *reinterpret_cast<const uint4*>(g);
#endif
}

template <int N>
__device__ __forceinline__ void wait_stage() {
#if USE_ASYNC
#if __has_builtin(__builtin_amdgcn_s_wait_asynccnt)
  __builtin_amdgcn_s_wait_asynccnt(N);
#else
  asm volatile("s_wait_asynccnt %0" :: "i"(N) : "memory");
#endif
#endif
}

// A-fragment (16x32 bf16), A[m][k] = base[m*stride + k]  (row-major, contiguous).
__device__ __forceinline__ v16bf load_a(const unsigned short* base, int stride) {
  const int lane = threadIdx.x & 31;
  const int m  = lane & 15;
  const int kb = (lane >> 4) << 4;
  const unsigned short* row = base + (size_t)m * stride + kb;
  BF16x16 f;
#pragma unroll
  for (int j = 0; j < 8; ++j) f.u[j]     = row[j];
#pragma unroll
  for (int j = 0; j < 8; ++j) f.u[8 + j] = row[8 + j];
  return f.v;
}

// B-fragment (32x16 bf16), K-major storage: B[k][n] = base[n*stride + k].
__device__ __forceinline__ v16bf load_bk(const unsigned short* base, int stride) {
  const int lane = threadIdx.x & 31;
  const int n  = lane & 15;
  const int kb = (lane >> 4) << 4;
  const unsigned short* row = base + (size_t)n * stride + kb;
  BF16x16 f;
#pragma unroll
  for (int j = 0; j < 16; ++j) f.u[j] = row[j];
  return f.v;
}

// ---------------------------------------------------------------------------
// Pre-pass A: x -> bf16, straight copy layout. 8 elements/thread.
// ---------------------------------------------------------------------------
__global__ __launch_bounds__(256) void k_xconv(const float* __restrict__ x,
                                               unsigned short* __restrict__ xb) {
  const size_t i = ((size_t)blockIdx.x * 256 + threadIdx.x) * 8;
  const float4 f0 = *reinterpret_cast<const float4*>(x + i);
  const float4 f1 = *reinterpret_cast<const float4*>(x + i + 4);
  unsigned short pk[8] = {f2bf(f0.x), f2bf(f0.y), f2bf(f0.z), f2bf(f0.w),
                          f2bf(f1.x), f2bf(f1.y), f2bf(f1.z), f2bf(f1.w)};
  *reinterpret_cast<uint4*>(xb + i) = *reinterpret_cast<const uint4*>(pk);
}

// ---------------------------------------------------------------------------
// Pre-pass B: weights -> bf16, K-major [col][k] (LDS-ready), via 32x32 LDS
// tile transpose so both global sides are coalesced.
// z=0: [W_Q|W_K|W_V] -> WB (3072 cols); z=1: W_O -> WOB (2048 cols).
// ---------------------------------------------------------------------------
__global__ __launch_bounds__(256) void k_wtrans(
    const float* __restrict__ Wq, const float* __restrict__ Wk, const float* __restrict__ Wv,
    const float* __restrict__ Wo, unsigned short* __restrict__ WB, unsigned short* __restrict__ WOB)
{
  __shared__ unsigned short t[32][33];
  const int k0 = blockIdx.x * 32;
  const int c0 = blockIdx.y * 32;
  const bool isWo = (blockIdx.z == 1);
  if (isWo && c0 >= DM) return;
  const int lc = threadIdx.x & 31;
  const int lr = threadIdx.x >> 5;          // 0..7

  // read (k, col), coalesced over col; element k at src[k*stride]
  const int col = c0 + lc;
  const float* src; int stride;
  if (isWo)            { src = Wo + col; stride = DM; }
  else if (col < 2048) { src = Wq + (size_t)(col >> 7) * DM * DH + (col & 127); stride = DH; }
  else if (col < 2560) { const int c = col - 2048; src = Wk + (size_t)(c >> 7) * DM * DH + (c & 127); stride = DH; }
  else                 { const int c = col - 2560; src = Wv + (size_t)(c >> 7) * DM * DH + (c & 127); stride = DH; }
#pragma unroll
  for (int i = 0; i < 4; ++i) {
    const int k = lr + i * 8;
    t[k][lc] = f2bf(src[(size_t)(k0 + k) * stride]);
  }
  __syncthreads();
  // write [col][k], coalesced over k
  unsigned short* dst = isWo ? WOB : WB;
#pragma unroll
  for (int i = 0; i < 4; ++i) {
    const int cw = lr + i * 8;
    dst[(size_t)(c0 + cw) * DM + k0 + lc] = t[lc][cw];
  }
}

// ---------------------------------------------------------------------------
// Kernel 1: fused QKV projection, all-bf16 operands, double-buffered async
// staging. Q, K row-major; V transposed (Vt[h][pos]) with packed 16B stores.
// BM=BN=128, BK=32; 8 waves as 2(M) x 4(N), each wave 4x2 WMMA tiles.
// ---------------------------------------------------------------------------
__global__ __launch_bounds__(256) void k_qkv(
    const unsigned short* __restrict__ xb, const unsigned short* __restrict__ WB,
    const float* __restrict__ bq, const float* __restrict__ bk, const float* __restrict__ bv,
    unsigned short* __restrict__ Q, unsigned short* __restrict__ K, unsigned short* __restrict__ Vt)
{
  __shared__ unsigned short As[2][128 * 40];
  __shared__ unsigned short Bs[2][128 * 40];
  const int tid  = threadIdx.x;
  const int lane = tid & 31, w = tid >> 5;
  const int wm = w & 1, wn = w >> 1;
  const int row0 = blockIdx.y * 128;
  const int col0 = blockIdx.x * 128;                 // never straddles Q/K/V
  const int lc = tid & 127;
  const int cg = col0 + lc;

  const int ar = tid >> 1, ac = (tid & 1) * 16;      // A staging map
  const int half = tid >> 7;                         // B staging map
  auto stage = [&](int buf, int k0) {
    const size_t asrc = (size_t)(row0 + ar) * DM + k0 + ac;
    cp128(&As[buf][ar * 40 + ac],     &xb[asrc]);
    cp128(&As[buf][ar * 40 + ac + 8], &xb[asrc + 8]);
    const size_t bsrc = (size_t)cg * DM + k0 + half * 16;
    cp128(&Bs[buf][lc * 40 + half * 16],     &WB[bsrc]);
    cp128(&Bs[buf][lc * 40 + half * 16 + 8], &WB[bsrc + 8]);
  };

  v8f acc[4][2] = {};
  stage(0, 0);
  for (int it = 0; it < DM / 32; ++it) {
    if (it + 1 < DM / 32) { stage((it + 1) & 1, (it + 1) * 32); wait_stage<4>(); }
    else                  { wait_stage<0>(); }
    __syncthreads();
    const unsigned short* Ab = As[it & 1];
    const unsigned short* Bb = Bs[it & 1];
#pragma unroll
    for (int mt = 0; mt < 4; ++mt) {
      const v16bf a = load_a(&Ab[(wm * 64 + mt * 16) * 40], 40);
#pragma unroll
      for (int nt = 0; nt < 2; ++nt) {
        const v16bf b = load_bk(&Bb[(wn * 32 + nt * 16) * 40], 40);
        acc[mt][nt] = __builtin_amdgcn_wmma_f32_16x16x32_bf16(
            false, a, false, b, (short)0, acc[mt][nt], false, false);
      }
    }
    __syncthreads();
  }

  const int n = lane & 15, mb = (lane >> 4) * 8;
#pragma unroll
  for (int mt = 0; mt < 4; ++mt)
#pragma unroll
    for (int nt = 0; nt < 2; ++nt) {
      const int cc = col0 + wn * 32 + nt * 16 + n;
      const int rr0 = row0 + wm * 64 + mt * 16 + mb;
      if (cc < 2048) {
        const float bias = bq[cc];
#pragma unroll
        for (int r = 0; r < 8; ++r)
          Q[(size_t)(rr0 + r) * DM + cc] = f2bf(acc[mt][nt][r] + bias);
      } else if (cc < 2560) {
        const int cl = cc - 2048;
        const float bias = bk[cl];
#pragma unroll
        for (int r = 0; r < 8; ++r)
          K[(size_t)(rr0 + r) * DKV + cl] = f2bf(acc[mt][nt][r] + bias);
      } else {
        const int cl = cc - 2560;
        const float bias = bv[cl];
        unsigned short pk[8];
#pragma unroll
        for (int r = 0; r < 8; ++r) pk[r] = f2bf(acc[mt][nt][r] + bias);
        *reinterpret_cast<uint4*>(&Vt[(size_t)cl * SEQ + rr0]) =
            *reinterpret_cast<const uint4*>(pk);
      }
    }
}

// ---------------------------------------------------------------------------
// Kernel 2: rotary embedding on Q (16 heads) and K (4 heads), in place, bf16.
// ---------------------------------------------------------------------------
__global__ void k_rotary(unsigned short* __restrict__ Q, unsigned short* __restrict__ K)
{
  const int idx = blockIdx.x * blockDim.x + threadIdx.x;
  const int qpairs = SEQ * (DM / 2);
  unsigned short* p; int pos, i;
  if (idx < qpairs) {
    pos = idx >> 10;
    const int rem = idx & 1023;
    i = rem & 63;
    p = Q + (size_t)pos * DM + (rem << 1);
  } else {
    const int id2 = idx - qpairs;
    if (id2 >= SEQ * (DKV / 2)) return;
    pos = id2 >> 8;
    const int rem = id2 & 255;
    i = rem & 63;
    p = K + (size_t)pos * DKV + (rem << 1);
  }
  const float freq = __powf(10000.0f, -(float)i * (1.0f / 64.0f));
  float s, c;
  __sincosf((float)pos * freq, &s, &c);
  const float x1 = bf2f(p[0]), x2 = bf2f(p[1]);
  p[0] = f2bf(x1 * c - x2 * s);
  p[1] = f2bf(x1 * s + x2 * c);
}

// ---------------------------------------------------------------------------
// Kernel 3: causal flash attention. Block = (head, 128-query tile); 8 waves x
// 16 rows; K (row-major) and Vt (feature-major) double-buffered through LDS
// in 64-key blocks via async copies; online softmax; contiguous b128 frags.
// ---------------------------------------------------------------------------
__global__ __launch_bounds__(256) void k_attn(
    const unsigned short* __restrict__ Q, const unsigned short* __restrict__ K,
    const unsigned short* __restrict__ Vt, unsigned short* __restrict__ Z)
{
  __shared__ unsigned short Ks[2][64 * 136];   // [key][h]  -> S B-frags contiguous
  __shared__ unsigned short Vs[2][128 * 72];   // [h][key]  -> PV B-frags contiguous
  __shared__ unsigned short Ps[8 * 16 * 72];   // per-wave P tiles, [m][key]
  const int tid  = threadIdx.x, lane = tid & 31, w = tid >> 5;
  const int qb = blockIdx.x, head = blockIdx.y, kvh = head >> 2;
  const int qrow = qb * 128 + w * 16;
  const int n = lane & 15, mb = (lane >> 4) * 8;
  const float inv_scale = 1.0f / 11.313708498984761f;

  const int rr = tid >> 2, cc4 = (tid & 3) * 32;   // K staging map
  const int hr = tid >> 1, hc  = (tid & 1) * 32;   // V staging map
  auto stageKV = [&](int buf, int j) {
    const size_t ksrc = (size_t)(j * 64 + rr) * DKV + kvh * DH + cc4;
#pragma unroll
    for (int i = 0; i < 4; ++i)
      cp128(&Ks[buf][rr * 136 + cc4 + i * 8], &K[ksrc + i * 8]);
    const size_t vsrc = (size_t)(kvh * DH + hr) * SEQ + j * 64 + hc;
#pragma unroll
    for (int i = 0; i < 4; ++i)
      cp128(&Vs[buf][hr * 72 + hc + i * 8], &Vt[vsrc + i * 8]);
  };

  v16bf qf[4];
#pragma unroll
  for (int kc = 0; kc < 4; ++kc)
    qf[kc] = load_a(Q + (size_t)qrow * DM + head * DH + kc * 32, DM);

  v8f o[8] = {};
  float mrow[8], lrow[8];
#pragma unroll
  for (int r = 0; r < 8; ++r) { mrow[r] = -3.0e38f; lrow[r] = 0.0f; }

  const int nkb = 2 * qb + 2;
  stageKV(0, 0);
  for (int j = 0; j < nkb; ++j) {
    if (j + 1 < nkb) { stageKV((j + 1) & 1, j + 1); wait_stage<8>(); }
    else             { wait_stage<0>(); }
    __syncthreads();
    const unsigned short* Kb = Ks[j & 1];
    const unsigned short* Vb = Vs[j & 1];

    // S = Q * K^T  (16 queries x 64 keys per wave); B[k][n]=Kb[n][k]
    v8f sfr[4] = {};
#pragma unroll
    for (int kc = 0; kc < 4; ++kc)
#pragma unroll
      for (int nt = 0; nt < 4; ++nt) {
        const v16bf b = load_bk(&Kb[(nt * 16) * 136 + kc * 32], 136);
        sfr[nt] = __builtin_amdgcn_wmma_f32_16x16x32_bf16(
            false, qf[kc], false, b, (short)0, sfr[nt], false, false);
      }

    // scale + causal mask (only last two key blocks touch the diagonal)
    const bool needmask = (j >= 2 * qb);
#pragma unroll
    for (int nt = 0; nt < 4; ++nt)
#pragma unroll
      for (int r = 0; r < 8; ++r) {
        float v = sfr[nt][r] * inv_scale;
        if (needmask && (j * 64 + nt * 16 + n) > (qrow + mb + r)) v = -3.0e38f;
        sfr[nt][r] = v;
      }

    // online softmax (16-lane butterflies; a row lives in one half-wave)
    float alpha[8], rmax[8], rsum[8];
#pragma unroll
    for (int r = 0; r < 8; ++r) {
      float v = fmaxf(fmaxf(sfr[0][r], sfr[1][r]), fmaxf(sfr[2][r], sfr[3][r]));
      v = fmaxf(v, __shfl_xor(v, 1, 32));
      v = fmaxf(v, __shfl_xor(v, 2, 32));
      v = fmaxf(v, __shfl_xor(v, 4, 32));
      v = fmaxf(v, __shfl_xor(v, 8, 32));
      const float mnew = fmaxf(mrow[r], v);
      alpha[r] = __expf(mrow[r] - mnew);
      mrow[r] = mnew; rmax[r] = mnew; rsum[r] = 0.0f;
    }
    unsigned short* pw = &Ps[w * 16 * 72];
#pragma unroll
    for (int nt = 0; nt < 4; ++nt)
#pragma unroll
      for (int r = 0; r < 8; ++r) {
        const float p = __expf(sfr[nt][r] - rmax[r]);
        rsum[r] += p;
        pw[(mb + r) * 72 + nt * 16 + n] = f2bf(p);
      }
#pragma unroll
    for (int r = 0; r < 8; ++r) {
      float v = rsum[r];
      v += __shfl_xor(v, 1, 32);
      v += __shfl_xor(v, 2, 32);
      v += __shfl_xor(v, 4, 32);
      v += __shfl_xor(v, 8, 32);
      lrow[r] = lrow[r] * alpha[r] + v;
    }
#pragma unroll
    for (int t = 0; t < 8; ++t)
#pragma unroll
      for (int r = 0; r < 8; ++r) o[t][r] *= alpha[r];
    __syncthreads();

    // O += P * V; B[k][n] = Vb[n][k] (contiguous)
#pragma unroll
    for (int kc = 0; kc < 2; ++kc) {
      const v16bf a = load_a(pw + kc * 32, 72);
#pragma unroll
      for (int t = 0; t < 8; ++t) {
        const v16bf b = load_bk(&Vb[(t * 16) * 72 + kc * 32], 72);
        o[t] = __builtin_amdgcn_wmma_f32_16x16x32_bf16(
            false, a, false, b, (short)0, o[t], false, false);
      }
    }
    __syncthreads();
  }

  // finalize: O / l, write z (bf16) as [seq][head*128 + h]
#pragma unroll
  for (int t = 0; t < 8; ++t)
#pragma unroll
    for (int r = 0; r < 8; ++r) {
      const float v = o[t][r] / lrow[r];
      Z[(size_t)(qrow + mb + r) * DM + head * DH + t * 16 + n] = f2bf(v);
    }
}

// ---------------------------------------------------------------------------
// Kernel 4: output projection. out = Z @ WOB^T + b_O (WOB is K-major bf16).
// Same double-buffered async pipeline. f32 output.
// ---------------------------------------------------------------------------
__global__ __launch_bounds__(256) void k_oproj(
    const unsigned short* __restrict__ Z, const unsigned short* __restrict__ WOB,
    const float* __restrict__ bo, float* __restrict__ out)
{
  __shared__ unsigned short As[2][128 * 40];
  __shared__ unsigned short Bs[2][128 * 40];
  const int tid  = threadIdx.x;
  const int lane = tid & 31, w = tid >> 5;
  const int wm = w & 1, wn = w >> 1;
  const int row0 = blockIdx.y * 128;
  const int col0 = blockIdx.x * 128;
  const int lc = tid & 127;

  const int ar = tid >> 1, ac = (tid & 1) * 16;
  const int half = tid >> 7;
  auto stage = [&](int buf, int k0) {
    const size_t asrc = (size_t)(row0 + ar) * DM + k0 + ac;
    cp128(&As[buf][ar * 40 + ac],     &Z[asrc]);
    cp128(&As[buf][ar * 40 + ac + 8], &Z[asrc + 8]);
    const size_t bsrc = (size_t)(col0 + lc) * DM + k0 + half * 16;
    cp128(&Bs[buf][lc * 40 + half * 16],     &WOB[bsrc]);
    cp128(&Bs[buf][lc * 40 + half * 16 + 8], &WOB[bsrc + 8]);
  };

  v8f acc[4][2] = {};
  stage(0, 0);
  for (int it = 0; it < DM / 32; ++it) {
    if (it + 1 < DM / 32) { stage((it + 1) & 1, (it + 1) * 32); wait_stage<4>(); }
    else                  { wait_stage<0>(); }
    __syncthreads();
    const unsigned short* Ab = As[it & 1];
    const unsigned short* Bb = Bs[it & 1];
#pragma unroll
    for (int mt = 0; mt < 4; ++mt) {
      const v16bf a = load_a(&Ab[(wm * 64 + mt * 16) * 40], 40);
#pragma unroll
      for (int nt = 0; nt < 2; ++nt) {
        const v16bf b = load_bk(&Bb[(wn * 32 + nt * 16) * 40], 40);
        acc[mt][nt] = __builtin_amdgcn_wmma_f32_16x16x32_bf16(
            false, a, false, b, (short)0, acc[mt][nt], false, false);
      }
    }
    __syncthreads();
  }

  const int n = lane & 15, mb = (lane >> 4) * 8;
#pragma unroll
  for (int mt = 0; mt < 4; ++mt)
#pragma unroll
    for (int nt = 0; nt < 2; ++nt) {
      const int cc = col0 + wn * 32 + nt * 16 + n;
      const float bias = bo[cc];
#pragma unroll
      for (int r = 0; r < 8; ++r) {
        const int rr = row0 + wm * 64 + mt * 16 + mb + r;
        out[(size_t)rr * DM + cc] = acc[mt][nt][r] + bias;
      }
    }
}

// ---------------------------------------------------------------------------
extern "C" void kernel_launch(void* const* d_in, const int* in_sizes, int n_in,
                              void* d_out, int out_size, void* d_ws, size_t ws_size,
                              hipStream_t stream) {
  const float* x  = (const float*)d_in[0];
  const float* Wq = (const float*)d_in[1];
  const float* Wk = (const float*)d_in[2];
  const float* Wv = (const float*)d_in[3];
  const float* Wo = (const float*)d_in[4];
  const float* bq = (const float*)d_in[5];
  const float* bk = (const float*)d_in[6];
  const float* bv = (const float*)d_in[7];
  const float* bo = (const float*)d_in[8];
  float* out = (float*)d_out;

  // Workspace (bf16): xb 16.8 | WB 12.6 | WOB 8.4 | Q 16.8 | K 4.2 | Vt 4.2 | Z 16.8 MB
  char* ws = (char*)d_ws;
  size_t off = 0;
  unsigned short* xb  = (unsigned short*)(ws + off); off += (size_t)SEQ * DM * 2;
  unsigned short* WB  = (unsigned short*)(ws + off); off += (size_t)NCOLS * DM * 2;
  unsigned short* WOB = (unsigned short*)(ws + off); off += (size_t)DM * DM * 2;
  unsigned short* Q   = (unsigned short*)(ws + off); off += (size_t)SEQ * DM * 2;
  unsigned short* K   = (unsigned short*)(ws + off); off += (size_t)SEQ * DKV * 2;
  unsigned short* Vt  = (unsigned short*)(ws + off); off += (size_t)SEQ * DKV * 2;
  unsigned short* Z   = (unsigned short*)(ws + off);

  k_xconv<<<SEQ * DM / (256 * 8), 256, 0, stream>>>(x, xb);
  k_wtrans<<<dim3(DM / 32, NCOLS / 32, 2), 256, 0, stream>>>(Wq, Wk, Wv, Wo, WB, WOB);
  k_qkv<<<dim3(NCOLS / 128, SEQ / 128), 256, 0, stream>>>(xb, WB, bq, bk, bv, Q, K, Vt);
  const int npairs = SEQ * (DM / 2) + SEQ * (DKV / 2);
  k_rotary<<<(npairs + 255) / 256, 256, 0, stream>>>(Q, K);
  k_attn<<<dim3(SEQ / 128, 16), 256, 0, stream>>>(Q, K, Vt, Z);
  k_oproj<<<dim3(DM / 128, SEQ / 128), 256, 0, stream>>>(Z, WOB, bo, out);
}
